// SimplifiedDRN_62483184222677
// MI455X (gfx1250) — compile-verified, matrix operand order checked by reading, not run on hardware
//
#include <hip/hip_runtime.h>
#include <cstdint>
#include <cstddef>

#define DEVINL __device__ __forceinline__

typedef __attribute__((ext_vector_type(16))) __bf16 v16bf;
typedef __attribute__((ext_vector_type(8)))  float  v8f;

union FragAB { v16bf v; unsigned u[8]; };

DEVINL float bf2f(unsigned short h) {
  return __builtin_bit_cast(float, ((unsigned)h) << 16);
}
DEVINL unsigned short f2bf(float f) {
  unsigned u = __builtin_bit_cast(unsigned, f);
  u += 0x7FFFu + ((u >> 16) & 1u);          // round-to-nearest-even
  return (unsigned short)(u >> 16);
}
// packed bf16x2 multiply -- single VOP3P instruction, co-executes with XDL WMMA
DEVINL unsigned pk_mul_bf16(unsigned a, unsigned b) {
  unsigned d;
  asm("v_pk_mul_bf16 %0, %1, %2" : "=v"(d) : "v"(a), "v"(b));
  return d;
}
// fragment = two contiguous 16B LDS reads (ISA 16-bit A/B layout: VGPR0-3 hold
// k=8*half..8*half+7, VGPR4-7 hold k=16+8*half..16+8*half+7)
DEVINL void load_frag(const unsigned short* p, FragAB& f) {
  const uint4 lo = *(const uint4*)p;         // k : base .. base+7
  const uint4 hi = *(const uint4*)(p + 16);  // k : base+16 .. base+23
  f.u[0] = lo.x; f.u[1] = lo.y; f.u[2] = lo.z; f.u[3] = lo.w;
  f.u[4] = hi.x; f.u[5] = hi.y; f.u[6] = hi.z; f.u[7] = hi.w;
}

// ---------------------------------------------------------------------------
// fp32 -> bf16 elementwise (x activations)
// ---------------------------------------------------------------------------
__global__ __launch_bounds__(256) void f32_to_bf16_kernel(
    const float* __restrict__ in, unsigned short* __restrict__ out, int n4)
{
  const int t = blockIdx.x * blockDim.x + threadIdx.x;
  if (t >= n4) return;
  const size_t i = (size_t)t * 4;
  const float4 v = *(const float4*)(in + i);
  uint2 o;
  o.x = (unsigned)f2bf(v.x) | ((unsigned)f2bf(v.y) << 16);
  o.y = (unsigned)f2bf(v.z) | ((unsigned)f2bf(v.w) << 16);
  *(uint2*)(out + i) = o;
}

// ---------------------------------------------------------------------------
// W[p][K][N] fp32 -> Wt[p][N][K] bf16, LDS-tiled transpose (coalesced both ways)
// ---------------------------------------------------------------------------
__global__ __launch_bounds__(256) void transpose_convert_kernel(
    const float* __restrict__ W, unsigned short* __restrict__ Wt, int K, int N)
{
  __shared__ unsigned short tile[32][33];
  const int p = blockIdx.z;
  const float* Wp = W + (size_t)p * K * N;
  unsigned short* Wtp = Wt + (size_t)p * N * K;
  const int k0 = blockIdx.x * 32, n0 = blockIdx.y * 32;
  const int tx = threadIdx.x, ty = threadIdx.y;
#pragma unroll
  for (int i = 0; i < 32; i += 8)
    tile[ty + i][tx] = f2bf(Wp[(size_t)(k0 + ty + i) * N + n0 + tx]);
  __syncthreads();
#pragma unroll
  for (int i = 0; i < 32; i += 8)
    Wtp[(size_t)(n0 + ty + i) * K + k0 + tx] = tile[tx][ty + i];
}

// ---------------------------------------------------------------------------
// selector: probs = softmax(X @ S + sb) over P=8 experts, one wave32 per row
// ---------------------------------------------------------------------------
__global__ __launch_bounds__(256) void selector_kernel(
    const unsigned short* __restrict__ X, const float* __restrict__ S,
    const float* __restrict__ sb, float* __restrict__ probs, int K)
{
  const int gw   = (int)((blockIdx.x * blockDim.x + threadIdx.x) >> 5);
  const int lane = threadIdx.x & 31;
  const unsigned short* row = X + (size_t)gw * K;
  float acc[8] = {0.f,0.f,0.f,0.f,0.f,0.f,0.f,0.f};
  for (int k = lane; k < K; k += 32) {
    const float xv = bf2f(row[k]);
    const float* sr = S + (size_t)k * 8;
#pragma unroll
    for (int p = 0; p < 8; ++p) acc[p] += xv * sr[p];
  }
#pragma unroll
  for (int off = 16; off > 0; off >>= 1) {
#pragma unroll
    for (int p = 0; p < 8; ++p) acc[p] += __shfl_xor(acc[p], off, 32);
  }
  float z[8];
#pragma unroll
  for (int p = 0; p < 8; ++p) z[p] = acc[p] + sb[p];
  float m = z[0];
#pragma unroll
  for (int p = 1; p < 8; ++p) m = fmaxf(m, z[p]);
  float s = 0.f;
#pragma unroll
  for (int p = 0; p < 8; ++p) { z[p] = __expf(z[p] - m); s += z[p]; }
  if (lane < 8) probs[(size_t)gw * 8 + lane] = z[lane] / s;
}

// ---------------------------------------------------------------------------
// Fused soft-MoE layer GEMM:
//   out[b,n] = act( sum_p probs[b,p] * ( sum_k A[b,k]*Wt[p][n][k] + bias[p,n] ) )
// Routing prob folded into A fragments (v_pk_mul_bf16) -> single accumulator,
// K_eff = P*K. Block tile 128x128, 8 waves (4M x 2N), wave tile 32x64,
// bf16 WMMA 16x16x32. A k-slice staged in LDS once, reused by all experts.
// Per expert: issue all 8 ds_load_b128 for B-frags first, overlap with the
// (LDS-independent) pk_mul A-scaling, then run the 8 WMMAs -> one wait point.
// ---------------------------------------------------------------------------
template<int P, bool RELU, bool STOREBF>
__global__ __launch_bounds__(256) void drn_gemm_kernel(
    const unsigned short* __restrict__ A,    // [B,K] bf16
    const unsigned short* __restrict__ Wt,   // [P,N,K] bf16
    const float* __restrict__ bias,          // [P,N] (or [N] if P==1)
    const float* __restrict__ probs,         // [B,P] or null
    unsigned short* __restrict__ outB,       // bf16 out (hidden layers)
    float* __restrict__ outF,                // fp32 out (classifier)
    int K, int N)
{
  constexpr int BM = 128, BN = 128, KT = 32, LDT = 40;   // 40 shorts: 16B rows, bank-safe
  constexpr bool HP  = (P > 1);
  constexpr int  PG  = HP ? 4 : 1;                       // experts per LDS phase
  constexpr int  NPH = P / PG;

  __shared__ unsigned short As[BM][LDT];          // 10.0 KB
  __shared__ unsigned short Bs[PG][BN][LDT];      // 40.0 KB (P=8) / 10 KB (P=1)

  const int tid  = threadIdx.x;
  const int lane = tid & 31;
  const int wv   = tid >> 5;
  const int lh   = lane >> 4;        // half-wave
  const int idx  = lane & 15;
  const int wm   = (wv & 3) * 32;    // wave M offset (4 waves over M)
  const int wn   = (wv >> 2) * 64;   // wave N offset (2 waves over N)
  const int mBlock = blockIdx.y * BM;
  const int nBlock = blockIdx.x * BN;

  v8f acc[2][4];
  const v8f vz = {0.f,0.f,0.f,0.f,0.f,0.f,0.f,0.f};
#pragma unroll
  for (int i = 0; i < 2; ++i)
#pragma unroll
    for (int j = 0; j < 4; ++j) acc[i][j] = vz;

  // routing probs for this lane's A-fragment rows, packed as bf16x2
  unsigned prpk[2][HP ? P : 1];
  if constexpr (HP) {
#pragma unroll
    for (int i = 0; i < 2; ++i) {
      const int row = mBlock + wm + i * 16 + idx;
#pragma unroll
      for (int p = 0; p < P; ++p)
        prpk[i][p] = (unsigned)f2bf(probs[(size_t)row * P + p]) * 0x00010001u;
    }
  }

  const int stR = tid >> 2;          // 64 rows per staging round
  const int stC = (tid & 3) * 8;     // 16B chunk within a 32-wide k-slice

  for (int kt = 0; kt < K; kt += KT) {
    FragAB araw[2];
#pragma unroll
    for (int ph = 0; ph < NPH; ++ph) {
      __syncthreads();
      if (ph == 0) {                 // stage A k-slice once; reused for all experts
#pragma unroll
        for (int rr = 0; rr < BM / 64; ++rr) {
          const int r = rr * 64 + stR;
          const unsigned short* gp = A + (size_t)(mBlock + r) * K + kt + stC;
          *(uint4*)(&As[r][stC]) = *(const uint4*)gp;
          __builtin_prefetch(gp + KT, 0, 3);
        }
      }
#pragma unroll
      for (int rr = 0; rr < (PG * BN) / 64; ++rr) {   // stage PG experts' B tiles
        const int r  = rr * 64 + stR;
        const int pp = r >> 7;                        // / BN
        const int nn = r & (BN - 1);
        const unsigned short* gp =
            Wt + ((size_t)(ph * PG + pp) * N + nBlock + nn) * K + kt + stC;
        *(uint4*)(&Bs[pp][nn][stC]) = *(const uint4*)gp;
        __builtin_prefetch(gp + KT, 0, 3);
      }
      __syncthreads();
      if (ph == 0) {                 // raw A fragments: 2 x ds_load_b128 each
#pragma unroll
        for (int i = 0; i < 2; ++i)
          load_frag(&As[wm + i * 16 + idx][lh * 8], araw[i]);
      }
#pragma unroll
      for (int pp = 0; pp < PG; ++pp) {
        // 1) issue all B-fragment LDS loads for this expert (8 x ds_load_b128)
        FragAB bf[4];
#pragma unroll
        for (int j = 0; j < 4; ++j)
          load_frag(&Bs[pp][wn + j * 16 + idx][lh * 8], bf[j]);
        // 2) LDS-independent A scaling overlaps the in-flight LDS loads
        FragAB af[2];
#pragma unroll
        for (int i = 0; i < 2; ++i) {
          if constexpr (HP) {
            const unsigned s = prpk[i][ph * PG + pp];
#pragma unroll
            for (int v = 0; v < 8; ++v) af[i].u[v] = pk_mul_bf16(araw[i].u[v], s);
          } else {
            af[i] = araw[i];
          }
        }
        // 3) matrix core work
#pragma unroll
        for (int j = 0; j < 4; ++j)
#pragma unroll
          for (int i = 0; i < 2; ++i)
            acc[i][j] = __builtin_amdgcn_wmma_f32_16x16x32_bf16(
                false, af[i].v, false, bf[j].v, short(0), acc[i][j], false, false);
      }
    }
  }

  // ---- epilogue: + sum_p probs*bias, activation, store ----
  int colj[4];
#pragma unroll
  for (int j = 0; j < 4; ++j) colj[j] = nBlock + wn + j * 16 + idx;

  float bcol[4][HP ? P : 1];
#pragma unroll
  for (int j = 0; j < 4; ++j) {
    if constexpr (HP) {
#pragma unroll
      for (int p = 0; p < P; ++p) bcol[j][p] = bias[(size_t)p * N + colj[j]];
    } else {
      bcol[j][0] = bias[colj[j]];
    }
  }

#pragma unroll
  for (int i = 0; i < 2; ++i) {
#pragma unroll
    for (int r = 0; r < 8; ++r) {
      const int row = mBlock + wm + i * 16 + lh * 8 + r;   // C/D: M = r + 8*half
      float pv[HP ? P : 1];
      if constexpr (HP) {
#pragma unroll
        for (int p = 0; p < P; ++p) pv[p] = probs[(size_t)row * P + p];
      }
#pragma unroll
      for (int j = 0; j < 4; ++j) {
        float bm = 0.f;
        if constexpr (HP) {
#pragma unroll
          for (int p = 0; p < P; ++p) bm += pv[p] * bcol[j][p];
        } else {
          bm = bcol[j][0];
        }
        float val = acc[i][j][r] + bm;
        if constexpr (RELU) val = fmaxf(val, 0.f);
        const size_t o = (size_t)row * N + colj[j];
        if constexpr (STOREBF) outB[o] = f2bf(val);
        else                   outF[o] = val;
      }
    }
  }
}

// ---------------------------------------------------------------------------
extern "C" void kernel_launch(void* const* d_in, const int* in_sizes, int n_in,
                              void* d_out, int out_size, void* d_ws, size_t ws_size,
                              hipStream_t stream)
{
  constexpr int B = 8192, IN = 1024, H1 = 2048, H2 = 2048, OUT = 1024, P = 8;
  const float* x   = (const float*)d_in[0];
  const float* W1  = (const float*)d_in[1];
  const float* b1  = (const float*)d_in[2];
  const float* S1  = (const float*)d_in[3];
  const float* sb1 = (const float*)d_in[4];
  const float* W2  = (const float*)d_in[5];
  const float* b2  = (const float*)d_in[6];
  const float* S2  = (const float*)d_in[7];
  const float* sb2 = (const float*)d_in[8];
  const float* Wc  = (const float*)d_in[9];
  const float* bc  = (const float*)d_in[10];
  float* out = (float*)d_out;
  (void)in_sizes; (void)n_in; (void)out_size; (void)ws_size;

  char* ws = (char*)d_ws;
  size_t off = 0;
  auto alloc = [&](size_t bytes) -> void* {
    void* p = ws + off;
    off += (bytes + 255) & ~(size_t)255;
    return p;
  };
  unsigned short* xb  = (unsigned short*)alloc((size_t)B * IN * 2);       //  16 MB
  unsigned short* W1t = (unsigned short*)alloc((size_t)P * IN * H1 * 2);  //  32 MB
  unsigned short* W2t = (unsigned short*)alloc((size_t)P * H1 * H2 * 2);  //  64 MB
  unsigned short* Wct = (unsigned short*)alloc((size_t)H2 * OUT * 2);     //   4 MB
  unsigned short* h1  = (unsigned short*)alloc((size_t)B * H1 * 2);       //  32 MB
  unsigned short* h2  = (unsigned short*)alloc((size_t)B * H2 * 2);       //  32 MB
  float* probs1 = (float*)alloc((size_t)B * P * 4);
  float* probs2 = (float*)alloc((size_t)B * P * 4);

  // precision conversion + weight transpose (one-off, bandwidth-trivial)
  {
    const int n4 = (B * IN) / 4;
    f32_to_bf16_kernel<<<(n4 + 255) / 256, 256, 0, stream>>>(x, xb, n4);
  }
  transpose_convert_kernel<<<dim3(IN / 32, H1 / 32, P), dim3(32, 8), 0, stream>>>(W1, W1t, IN, H1);
  transpose_convert_kernel<<<dim3(H1 / 32, H2 / 32, P), dim3(32, 8), 0, stream>>>(W2, W2t, H1, H2);
  transpose_convert_kernel<<<dim3(H2 / 32, OUT / 32, 1), dim3(32, 8), 0, stream>>>(Wc, Wct, H2, OUT);

  // layer 1
  selector_kernel<<<B / 8, 256, 0, stream>>>(xb, S1, sb1, probs1, IN);
  drn_gemm_kernel<P, true, true><<<dim3(H1 / 128, B / 128), 256, 0, stream>>>(
      xb, W1t, b1, probs1, h1, nullptr, IN, H1);
  // layer 2
  selector_kernel<<<B / 8, 256, 0, stream>>>(h1, S2, sb2, probs2, H1);
  drn_gemm_kernel<P, true, true><<<dim3(H2 / 128, B / 128), 256, 0, stream>>>(
      h1, W2t, b2, probs2, h2, nullptr, H1, H2);
  // classifier (P=1, fp32 out, no relu)
  drn_gemm_kernel<1, false, false><<<dim3(OUT / 128, B / 128), 256, 0, stream>>>(
      h2, Wct, bc, nullptr, nullptr, out, H2, OUT);
}
